// AGNNet_49959059587652
// MI455X (gfx1250) — matrix operand order repaired and checked by reading, not compiled, not run on hardware
//
#include <hip/hip_runtime.h>

typedef __attribute__((ext_vector_type(16))) __bf16 v16bf;
typedef __attribute__((ext_vector_type(8)))  float  v8f;

#define HD __device__ __forceinline__

constexpr int INdim = 256, Hdim = 256, Fdim = 512, OUTdim = 64, LAYERS = 3, TOPKK = 8;
constexpr float TAUv = 0.9f;

union Frag { v16bf v; uint4 q[2]; };

HD unsigned short f2bf(float f) {
  unsigned u = __float_as_uint(f);
  return (unsigned short)((u + 0x7FFFu + ((u >> 16) & 1u)) >> 16);
}

HD float wsum(float v) {
  for (int o = 16; o > 0; o >>= 1) v += __shfl_down(v, o);
  return v;
}

// ---------------- fp32 -> bf16 conversion ----------------
__global__ void cvt_bf16_kernel(const float* __restrict__ in,
                                unsigned short* __restrict__ out, long n) {
  long i = (long)blockIdx.x * blockDim.x + threadIdx.x;
  if (i < n) out[i] = f2bf(in[i]);
}

// ------------- pack fp32 KxN weight into WMMA B-fragment bf16 layout -------------
// fragment (nb, kb): 32 lanes x 16 bf16 contiguous; lane<16 -> n=nb*16+lane, k=kb*32+i
//                    lane>=16 -> n=nb*16+(lane-16), k=kb*32+16+i
__global__ void pack_b_kernel(const float* __restrict__ W,
                              unsigned short* __restrict__ P, int K, int N) {
  int idx = blockIdx.x * blockDim.x + threadIdx.x;
  if (idx >= K * N) return;
  int i = idx & 15;
  int lane = (idx >> 4) & 31;
  int f = idx >> 9;
  int KB = K >> 5;
  int kb = f % KB, nb = f / KB;
  int k = kb * 32 + ((lane >> 4) << 4) + i;
  int n = nb * 16 + (lane & 15);
  P[idx] = f2bf(W[(size_t)k * N + n]);
}

// ---------------- WMMA GEMM: C = act(A_bf16[MxK] @ Bpacked + bias) (+resid) ----------------
// block = 256 threads = 8 waves, tile 128(M) x 64(N), K stepped by 32
__global__ __launch_bounds__(256)
void gemm_bf16_kernel(const unsigned short* __restrict__ A,
                      const unsigned short* __restrict__ BP,
                      const float* __restrict__ bias,
                      const float* __restrict__ resid,
                      float* __restrict__ outF,
                      unsigned short* __restrict__ outB,
                      int M, int K, int N, int act) {
  __shared__ __align__(16) unsigned short As[128][40];  // 32 k-elems + pad
  const int tid = threadIdx.x;
  const int lane = tid & 31;
  const int wave = tid >> 5;
  const int wm = wave & 3, wn = wave >> 2;   // 4x2 wave grid of 32x32 tiles
  const int bm = blockIdx.y * 128;
  const int bn = blockIdx.x * 64;
  const int KB = K >> 5;

  v8f c[2][2] = {};

  const int row_st = tid >> 1;   // 0..127
  const int seg = tid & 1;       // two 32B segments per 64B row

  for (int kb = 0; kb < KB; ++kb) {
    // stage 128x32 bf16 A tile into LDS (coalesced b128 loads)
    {
      int gr = bm + row_st;
      uint4 z0 = {0u, 0u, 0u, 0u}, z1 = {0u, 0u, 0u, 0u};
      if (gr < M) {
        const uint4* s = (const uint4*)(A + (size_t)gr * K + kb * 32 + seg * 16);
        z0 = s[0]; z1 = s[1];
      }
      *(uint4*)&As[row_st][seg * 16] = z0;
      *(uint4*)&As[row_st][seg * 16 + 8] = z1;
    }
    __syncthreads();

    // A fragments (contiguous k = 16*half + i, matching packed B): half = lane>>4
    Frag a0, a1;
    {
      int half = lane >> 4;
      int m0 = wm * 32 + (lane & 15);
      a0.q[0] = *(const uint4*)&As[m0][half * 16];
      a0.q[1] = *(const uint4*)&As[m0][half * 16 + 8];
      int m1 = m0 + 16;
      a1.q[0] = *(const uint4*)&As[m1][half * 16];
      a1.q[1] = *(const uint4*)&As[m1][half * 16 + 8];
    }

#pragma unroll
    for (int tn = 0; tn < 2; ++tn) {
      Frag b;
      int n16 = (bn >> 4) + wn * 2 + tn;
      const uint4* bp = (const uint4*)(BP + ((size_t)(n16 * KB + kb) * 32 + lane) * 16);
      b.q[0] = bp[0]; b.q[1] = bp[1];
      c[0][tn] = __builtin_amdgcn_wmma_f32_16x16x32_bf16(
          false, a0.v, false, b.v, (short)0, c[0][tn], false, false);
      c[1][tn] = __builtin_amdgcn_wmma_f32_16x16x32_bf16(
          false, a1.v, false, b.v, (short)0, c[1][tn], false, false);
    }
    __syncthreads();
  }

  // epilogue: C/D layout -> lane<16: (M=r, N=lane); lane>=16: (M=8+r, N=lane-16)
  int rbase = (lane >= 16) ? 8 : 0;
  int coll = lane & 15;
#pragma unroll
  for (int tm = 0; tm < 2; ++tm) {
#pragma unroll
    for (int tn = 0; tn < 2; ++tn) {
      int col = bn + wn * 32 + tn * 16 + coll;
#pragma unroll
      for (int r = 0; r < 8; ++r) {
        int row = bm + wm * 32 + tm * 16 + rbase + r;
        if (row < M) {
          float v = c[tm][tn][r] + bias[col];
          if (act == 1) v = fmaxf(v, 0.f);
          else if (act == 2) v = 0.5f * v * (1.f + erff(v * 0.70710678118654752f));
          if (resid) v += resid[(size_t)row * N + col];
          if (outF) outF[(size_t)row * N + col] = v;
          if (outB) outB[(size_t)row * N + col] = f2bf(v);
        }
      }
    }
  }
}

// ---------------- per-row stats: sum|h|, h.wp, h.Wa[:H], h.Wa[H:2H] ----------------
__global__ __launch_bounds__(256)
void rowstats_kernel(const float* __restrict__ h, const float* __restrict__ wp,
                     const float* __restrict__ Wa, float* delta, float* hdot,
                     float* adst, float* asrc, int Nn) {
  int lane = threadIdx.x & 31;
  int row = blockIdx.x * 8 + (threadIdx.x >> 5);
  if (row >= Nn) return;
  const float* hr = h + (size_t)row * Hdim;
  float sA = 0, sP = 0, sD = 0, sS = 0;
  for (int c = lane; c < Hdim; c += 32) {
    float v = hr[c];
    sA += fabsf(v);
    sP += v * wp[c];
    sD += v * Wa[c];
    sS += v * Wa[Hdim + c];
  }
  sA = wsum(sA); sP = wsum(sP); sD = wsum(sD); sS = wsum(sS);
  if (lane == 0) { delta[row] = sA; hdot[row] = sP; adst[row] = sD; asrc[row] = sS; }
}

__global__ void neigh_kernel(const int* __restrict__ src, const int* __restrict__ dst,
                             const float* __restrict__ delta, float* neigh, int E) {
  int e = blockIdx.x * blockDim.x + threadIdx.x;
  if (e < E) atomicAdd(&neigh[dst[e]], delta[src[e]]);
}

__global__ void pi_kernel(const float* __restrict__ hdot, const float* __restrict__ neigh,
                          float* pi, int Nn) {
  int i = blockIdx.x * blockDim.x + threadIdx.x;
  if (i < Nn) pi[i] = 1.f / (1.f + expf(-(hdot[i] + neigh[i])));
}

// ---------------- edge logits -> exp, denom, per-dst counts ----------------
__global__ void edge_kernel(const int* __restrict__ src, const int* __restrict__ dst,
                            const float* __restrict__ adst, const float* __restrict__ asrc,
                            const float* __restrict__ pi, const float* __restrict__ Wa,
                            const float* __restrict__ ba, float* __restrict__ exp_e,
                            float* denom, int* count, int E, int Nn) {
  int e = blockIdx.x * blockDim.x + threadIdx.x;
  int Et = E + Nn;
  if (e >= Et) return;
  int s = (e < E) ? src[e] : (e - E);
  int d = (e < E) ? dst[e] : (e - E);
  float v = adst[d] + asrc[s] + pi[s] * Wa[2 * Hdim] + ba[0];
  v = (v < 0.f) ? 0.2f * v : v;       // leaky relu 0.2
  v *= (1.0f / TAUv);
  v = fminf(fmaxf(v, -5.f), 5.f);
  float ex = expf(v);
  exp_e[e] = ex;
  atomicAdd(&denom[d], ex);
  atomicAdd(&count[d], 1);
}

// ---------------- single-block exclusive scan (n <= a few 100k) ----------------
__global__ void exscan_kernel(const int* __restrict__ in, int* __restrict__ out, int n) {
  __shared__ int tmp[1024];
  __shared__ int carry;
  if (threadIdx.x == 0) carry = 0;
  __syncthreads();
  for (int base = 0; base < n; base += 1024) {
    int i = base + threadIdx.x;
    int v = (i < n) ? in[i] : 0;
    tmp[threadIdx.x] = v;
    __syncthreads();
    for (int off = 1; off < 1024; off <<= 1) {
      int t = (threadIdx.x >= off) ? tmp[threadIdx.x - off] : 0;
      __syncthreads();
      tmp[threadIdx.x] += t;
      __syncthreads();
    }
    if (i < n) out[i] = carry + tmp[threadIdx.x] - v;
    __syncthreads();
    if (threadIdx.x == 0) carry += tmp[1023];
    __syncthreads();
  }
}

__global__ void scatter_kernel(const int* __restrict__ src, const int* __restrict__ dst,
                               const int* __restrict__ off, int* cursor,
                               int* __restrict__ eid, int E, int Nn) {
  int e = blockIdx.x * blockDim.x + threadIdx.x;
  int Et = E + Nn;
  if (e >= Et) return;
  int d = (e < E) ? dst[e] : (e - E);
  int p = off[d] + atomicAdd(&cursor[d], 1);
  eid[p] = e;
}

// ---------------- per-dst top-8 (rank with stable tie-break) + renorm ----------------
__global__ void topk_kernel(const int* __restrict__ off, const int* __restrict__ eid,
                            const float* __restrict__ exp_e, const float* __restrict__ denom,
                            float* __restrict__ alpha, int Nn) {
  int d = blockIdx.x * blockDim.x + threadIdx.x;
  if (d >= Nn) return;
  int s = off[d], e2 = off[d + 1];
  float inv = 1.f / (denom[d] + 1e-16f);
  float ksum = 0.f;
  for (int i = s; i < e2; ++i) {
    int ei = eid[i]; float ai = exp_e[ei];
    int rank = 0;
    for (int j = s; j < e2; ++j) {
      if (j == i) continue;
      int ej = eid[j]; float aj = exp_e[ej];
      if (aj > ai || (aj == ai && ej < ei)) rank++;
    }
    if (rank < TOPKK) ksum += ai * inv;
  }
  float rnorm = 1.f / (ksum + 1e-16f);
  for (int i = s; i < e2; ++i) {
    int ei = eid[i]; float ai = exp_e[ei];
    int rank = 0;
    for (int j = s; j < e2; ++j) {
      if (j == i) continue;
      int ej = eid[j]; float aj = exp_e[ej];
      if (aj > ai || (aj == ai && ej < ei)) rank++;
    }
    alpha[ei] = (rank < TOPKK) ? ai * inv * rnorm : 0.f;
  }
}

// ---------------- gather aggregate: h[d] += relu(sum alpha_e * xp[src_e]) ----------------
__global__ __launch_bounds__(256)
void agg_kernel(const int* __restrict__ src, const int* __restrict__ off,
                const int* __restrict__ eid, const float* __restrict__ alpha,
                const float* __restrict__ xp, float* __restrict__ h, int E, int Nn) {
  int lane = threadIdx.x & 31;
  int d = blockIdx.x * 8 + (threadIdx.x >> 5);
  if (d >= Nn) return;
  float acc[8] = {};
  int s0 = off[d], s1 = off[d + 1];
  for (int i = s0; i < s1; ++i) {
    int ei = eid[i];
    float a = alpha[ei];
    if (a == 0.f) continue;
    int sn = (ei < E) ? src[ei] : (ei - E);
    const float* xr = xp + (size_t)sn * Hdim;
#pragma unroll
    for (int k = 0; k < 8; ++k) acc[k] += a * xr[lane + 32 * k];
  }
  float* hr = h + (size_t)d * Hdim;
#pragma unroll
  for (int k = 0; k < 8; ++k) hr[lane + 32 * k] += fmaxf(acc[k], 0.f);
}

// ---------------- layernorm -> bf16 (input for next WMMA GEMM) ----------------
__global__ __launch_bounds__(256)
void ln_kernel(const float* __restrict__ h, const float* __restrict__ g,
               const float* __restrict__ b, unsigned short* __restrict__ out, int Nn) {
  int lane = threadIdx.x & 31;
  int row = blockIdx.x * 8 + (threadIdx.x >> 5);
  if (row >= Nn) return;
  const float* hr = h + (size_t)row * Hdim;
  float v[8]; float s = 0, s2 = 0;
#pragma unroll
  for (int k = 0; k < 8; ++k) {
    v[k] = hr[lane + 32 * k];
    s += v[k]; s2 += v[k] * v[k];
  }
  s = wsum(s); s2 = wsum(s2);
  s = __shfl(s, 0); s2 = __shfl(s2, 0);
  float mu = s * (1.f / Hdim);
  float var = s2 * (1.f / Hdim) - mu * mu;
  float inv = rsqrtf(var + 1e-5f);
  unsigned short* orow = out + (size_t)row * Hdim;
#pragma unroll
  for (int k = 0; k < 8; ++k) {
    int c = lane + 32 * k;
    orow[c] = f2bf((v[k] - mu) * inv * g[c] + b[c]);
  }
}

extern "C" void kernel_launch(void* const* d_in, const int* in_sizes, int n_in,
                              void* d_out, int out_size, void* d_ws, size_t ws_size,
                              hipStream_t stream) {
  const float* x    = (const float*)d_in[0];
  const int*   eix  = (const int*)d_in[1];
  const float* Wi   = (const float*)d_in[2];
  const float* bi   = (const float*)d_in[3];
  const float* wp   = (const float*)d_in[4];
  const float* Wa   = (const float*)d_in[5];
  const float* ba   = (const float*)d_in[6];
  const float* ln1g = (const float*)d_in[7];
  const float* ln1b = (const float*)d_in[8];
  const float* ln2g = (const float*)d_in[9];
  const float* ln2b = (const float*)d_in[10];
  const float* W1   = (const float*)d_in[11];
  const float* b1   = (const float*)d_in[12];
  const float* W2   = (const float*)d_in[13];
  const float* b2   = (const float*)d_in[14];
  const float* Wc   = (const float*)d_in[15];
  const float* bc   = (const float*)d_in[16];
  const float* Wo   = (const float*)d_in[17];
  const float* bo   = (const float*)d_in[18];

  const int Nn = in_sizes[0] / INdim;
  const int E  = in_sizes[1] / 2;
  const int Et = E + Nn;
  const int* srcI = eix;
  const int* dstI = eix + E;

  // ---- carve workspace ----
  char* p = (char*)d_ws;
  auto alloc = [&](size_t bytes) -> void* {
    void* r = (void*)p;
    p += (bytes + 255) & ~(size_t)255;
    return r;
  };
  float* h            = (float*)alloc((size_t)Nn * Hdim * 4);
  unsigned short* nbf = (unsigned short*)alloc((size_t)Nn * Hdim * 2); // bf16 GEMM A
  float* big          = (float*)alloc((size_t)Nn * Hdim * 4);          // xp fp32 / t bf16
  unsigned short* tbf = (unsigned short*)big;
  float* delta = (float*)alloc((size_t)Nn * 4);
  float* hdot  = (float*)alloc((size_t)Nn * 4);
  float* adstB = (float*)alloc((size_t)Nn * 4);
  float* asrcB = (float*)alloc((size_t)Nn * 4);
  float* piB   = (float*)alloc((size_t)Nn * 4);
  float* neigh = (float*)alloc((size_t)Nn * 4);
  float* denom = (float*)alloc((size_t)Nn * 4);
  float* expe  = (float*)alloc((size_t)Et * 4);
  float* alpha = (float*)alloc((size_t)Et * 4);
  int* cnt  = (int*)alloc((size_t)(Nn + 1) * 4);
  int* off  = (int*)alloc((size_t)(Nn + 1) * 4);
  int* cur  = (int*)alloc((size_t)(Nn + 1) * 4);
  int* eidb = (int*)alloc((size_t)Et * 4);
  unsigned short* WiP = (unsigned short*)alloc((size_t)INdim * Hdim * 2);
  unsigned short* WcP = (unsigned short*)alloc((size_t)LAYERS * Hdim * Hdim * 2);
  unsigned short* W1P = (unsigned short*)alloc((size_t)LAYERS * Hdim * Fdim * 2);
  unsigned short* W2P = (unsigned short*)alloc((size_t)LAYERS * Fdim * Hdim * 2);
  unsigned short* WoP = (unsigned short*)alloc((size_t)Hdim * OUTdim * 2);

  // ---- zero atomically-accumulated buffers (capture-safe) ----
  hipMemsetAsync(neigh, 0, (size_t)Nn * 4, stream);
  hipMemsetAsync(denom, 0, (size_t)Nn * 4, stream);
  hipMemsetAsync(cnt, 0, (size_t)(Nn + 1) * 4, stream);
  hipMemsetAsync(cur, 0, (size_t)(Nn + 1) * 4, stream);

  const int T = 256;
  // ---- pack weights into WMMA B layout ----
  auto packL = [&](const float* W, unsigned short* P, int K, int N) {
    int tot = K * N;
    pack_b_kernel<<<(tot + T - 1) / T, T, 0, stream>>>(W, P, K, N);
  };
  packL(Wi, WiP, INdim, Hdim);
  for (int l = 0; l < LAYERS; ++l) {
    packL(Wc + (size_t)l * Hdim * Hdim, WcP + (size_t)l * Hdim * Hdim, Hdim, Hdim);
    packL(W1 + (size_t)l * Hdim * Fdim, W1P + (size_t)l * Hdim * Fdim, Hdim, Fdim);
    packL(W2 + (size_t)l * Fdim * Hdim, W2P + (size_t)l * Fdim * Hdim, Fdim, Hdim);
  }
  packL(Wo, WoP, Hdim, OUTdim);

  // ---- x -> bf16, then h = relu(x @ Wi + bi) ----
  {
    long tot = (long)Nn * INdim;
    cvt_bf16_kernel<<<(int)((tot + T - 1) / T), T, 0, stream>>>(x, nbf, tot);
  }
  dim3 gB(Hdim / 64, (Nn + 127) / 128);
  gemm_bf16_kernel<<<gB, 256, 0, stream>>>(nbf, WiP, bi, nullptr, h, nullptr,
                                           Nn, INdim, Hdim, 1);

  // ---- priority / attention scaffolding ----
  rowstats_kernel<<<(Nn + 7) / 8, 256, 0, stream>>>(h, wp, Wa, delta, hdot, adstB, asrcB, Nn);
  neigh_kernel<<<(E + T - 1) / T, T, 0, stream>>>(srcI, dstI, delta, neigh, E);
  pi_kernel<<<(Nn + T - 1) / T, T, 0, stream>>>(hdot, neigh, piB, Nn);
  edge_kernel<<<(Et + T - 1) / T, T, 0, stream>>>(srcI, dstI, adstB, asrcB, piB, Wa, ba,
                                                  expe, denom, cnt, E, Nn);
  exscan_kernel<<<1, 1024, 0, stream>>>(cnt, off, Nn + 1);
  scatter_kernel<<<(Et + T - 1) / T, T, 0, stream>>>(srcI, dstI, off, cur, eidb, E, Nn);
  topk_kernel<<<(Nn + T - 1) / T, T, 0, stream>>>(off, eidb, expe, denom, alpha, Nn);

  // ---- layers ----
  float* xp = big;
  for (int l = 0; l < LAYERS; ++l) {
    ln_kernel<<<(Nn + 7) / 8, 256, 0, stream>>>(h, ln1g + l * Hdim, ln1b + l * Hdim, nbf, Nn);
    gemm_bf16_kernel<<<gB, 256, 0, stream>>>(nbf, WcP + (size_t)l * Hdim * Hdim,
                                             bc + l * Hdim, nullptr, xp, nullptr,
                                             Nn, Hdim, Hdim, 0);
    agg_kernel<<<(Nn + 7) / 8, 256, 0, stream>>>(srcI, off, eidb, alpha, xp, h, E, Nn);
    ln_kernel<<<(Nn + 7) / 8, 256, 0, stream>>>(h, ln2g + l * Hdim, ln2b + l * Hdim, nbf, Nn);
    dim3 g1(Fdim / 64, (Nn + 127) / 128);
    gemm_bf16_kernel<<<g1, 256, 0, stream>>>(nbf, W1P + (size_t)l * Hdim * Fdim,
                                             b1 + l * Fdim, nullptr, nullptr, tbf,
                                             Nn, Hdim, Fdim, 2);  // exact GELU
    gemm_bf16_kernel<<<gB, 256, 0, stream>>>(tbf, W2P + (size_t)l * Fdim * Hdim,
                                             b2 + l * Hdim, h, h, nullptr,
                                             Nn, Fdim, Hdim, 0);  // residual into h
  }

  // ---- out = h @ Wo + bo ----
  {
    long tot = (long)Nn * Hdim;
    cvt_bf16_kernel<<<(int)((tot + T - 1) / T), T, 0, stream>>>(h, nbf, tot);
  }
  dim3 gO(OUTdim / 64, (Nn + 127) / 128);
  gemm_bf16_kernel<<<gO, 256, 0, stream>>>(nbf, WoP, bo, nullptr, (float*)d_out, nullptr,
                                           Nn, Hdim, OUTdim, 0);
}
